// Adapter_59270548685191
// MI455X (gfx1250) — matrix operand order repaired and checked by reading, not compiled, not run on hardware
//
#include <hip/hip_runtime.h>

// ---------------- configuration ----------------
constexpr int DIMD = 768;   // model dim
constexpr int DIMR = 64;    // bottleneck
constexpr int WAVES = 8;
constexpr int TOK_PER_WAVE = 16;
constexpr int TOK_PER_BLOCK = WAVES * TOK_PER_WAVE;   // 128
constexpr int KFRAG_DOWN = DIMD / 32;                 // 24
constexpr int NTILE_DOWN = DIMR / 16;                 // 4
constexpr int KFRAG_UP   = DIMR / 32;                 // 2
constexpr int NTILE_UP   = DIMD / 16;                 // 48
constexpr float SCALE = 0.1f;
constexpr float LN_EPS = 1e-5f;

typedef __attribute__((ext_vector_type(16))) __bf16        v16bf;
typedef __attribute__((ext_vector_type(8)))  __bf16        v8bf;
typedef __attribute__((ext_vector_type(8)))  float         v8f;
typedef __attribute__((ext_vector_type(16))) float         f32x16;
typedef __attribute__((ext_vector_type(4)))  unsigned int  u32x4;

union FragU {
    v16bf           bf;
    u32x4           q[2];
    unsigned short  s[16];
};

union Bf8U {
    v8bf            bf;
    unsigned short  s[8];
};

// ---------------------------------------------------------------------------
// Pack fp32 weights into bf16, laid out in this file's per-lane WMMA B-fragment
// order (N = lane%16, K contiguous per lane) so the GEMM kernel does coalesced
// 32B-per-lane fragment loads.
//   Bd: [kT(24)][nT(4)] [lane(32)][16]   from W_down[768][64]
//   Bu: [kT(2) ][nT(48)][lane(32)][16]   from W_up  [64 ][768]
// ---------------------------------------------------------------------------
__global__ void pack_weights_kernel(const float* __restrict__ Wd,
                                    const float* __restrict__ Wu,
                                    unsigned short* __restrict__ Bd,
                                    unsigned short* __restrict__ Bu) {
    const int frag = blockIdx.x;          // 0..191
    const int lane = threadIdx.x;         // 0..31
    const int kadd = (lane < 16) ? 0 : 16;
    const int c16  = lane & 15;
    f32x16 w;
    unsigned short* dst;
    if (frag < KFRAG_DOWN * NTILE_DOWN) {
        const int kT = frag / NTILE_DOWN;
        const int nT = frag % NTILE_DOWN;
        const int kbase = kT * 32 + kadd;
        const int c = nT * 16 + c16;
        dst = Bd + ((size_t)frag * 32 + lane) * 16;
        #pragma unroll
        for (int j = 0; j < 16; ++j)
            w[j] = Wd[(size_t)(kbase + j) * DIMR + c];
    } else {
        const int f  = frag - KFRAG_DOWN * NTILE_DOWN;
        const int kT = f / NTILE_UP;
        const int nT = f % NTILE_UP;
        const int kbase = kT * 32 + kadd;
        const int c = nT * 16 + c16;
        dst = Bu + ((size_t)f * 32 + lane) * 16;
        #pragma unroll
        for (int j = 0; j < 16; ++j)
            w[j] = Wu[(size_t)(kbase + j) * DIMD + c];
    }
    FragU o;
    o.bf = __builtin_convertvector(w, v16bf);     // hw v_cvt_pk_bf16_f32
    *(u32x4*)(dst + 0) = o.q[0];
    *(u32x4*)(dst + 8) = o.q[1];
}

// ---------------------------------------------------------------------------
// Fused LayerNorm -> down-proj -> ReLU -> up-proj -> scale.
// One wave per 16-token M-tile; 8 waves / block.
// ---------------------------------------------------------------------------
__global__ __launch_bounds__(256) void adapter_kernel(
        const float* __restrict__ x,
        const float* __restrict__ gamma,
        const float* __restrict__ beta,
        const float* __restrict__ b_down,
        const float* __restrict__ b_up,
        const unsigned short* __restrict__ Bd,
        const unsigned short* __restrict__ Bu,
        float* __restrict__ out,
        int ntok) {
    __shared__ __align__(16) float          s_gamma[DIMD];
    __shared__ __align__(16) float          s_beta [DIMD];
    // wave-private bf16 staging tiles (padded row strides, 16B aligned)
    __shared__ __align__(16) unsigned short s_h[WAVES][16][40];  // 32 used/row
    __shared__ __align__(16) unsigned short s_z[WAVES][16][72];  // 64 used/row

    for (int i = threadIdx.x; i < DIMD; i += 256) {
        s_gamma[i] = gamma[i];
        s_beta[i]  = beta[i];
    }
    __syncthreads();                 // only cross-wave sync in the kernel

    const int wave = threadIdx.x >> 5;
    const int lane = threadIdx.x & 31;
    const int tokBase = blockIdx.x * TOK_PER_BLOCK + wave * TOK_PER_WAVE;
    if (tokBase >= ntok) return;     // wave-uniform: EXEC stays all-ones

    const float* xw = x + (size_t)tokBase * DIMD;

    // ---------------- Phase 1: LayerNorm statistics ----------------
    const int myTok = lane & 15;
    float myMu = 0.f, myRs = 0.f;
    for (int t = 0; t < 16; ++t) {
        const float4* row = (const float4*)(xw + (size_t)t * DIMD);
        float s = 0.f, s2 = 0.f;
        #pragma unroll
        for (int j = 0; j < DIMD / 128; ++j) {       // 6 float4 per lane
            float4 v = row[lane + 32 * j];
            s  += v.x + v.y + v.z + v.w;
            s2 += v.x * v.x + v.y * v.y + v.z * v.z + v.w * v.w;
        }
        #pragma unroll
        for (int off = 16; off > 0; off >>= 1) {
            s  += __shfl_xor(s,  off, 32);
            s2 += __shfl_xor(s2, off, 32);
        }
        float m   = s * (1.f / DIMD);
        float var = fmaxf(s2 * (1.f / DIMD) - m * m, 0.f);
        float r   = rsqrtf(var + LN_EPS);
        if (myTok == t) { myMu = m; myRs = r; }
    }

    // ---------------- Phase 2: normalize + down-proj WMMA ----------------
    v8f accD[NTILE_DOWN];
    #pragma unroll
    for (int n = 0; n < NTILE_DOWN; ++n) accD[n] = {};

    const int dh   = (lane >> 4) << 4;      // 0 / 16 : normalizer half-row
    const int aoff = (lane < 16) ? 0 : 16;  // A-fragment K offset
    const int arow = lane & 15;             // A-fragment row (M)

    for (int kc = 0; kc < KFRAG_DOWN; ++kc) {
        const int kbase = kc * 32;
        // each lane normalizes 16 dims of its token -> bf16 -> LDS
        const float4* xr = (const float4*)(xw + (size_t)myTok * DIMD + kbase + dh);
        f32x16 hv;
        #pragma unroll
        for (int q4 = 0; q4 < 4; ++q4) {
            float4 v = xr[q4];
            const int e = kbase + dh + q4 * 4;
            hv[q4 * 4 + 0] = (v.x - myMu) * myRs * s_gamma[e + 0] + s_beta[e + 0];
            hv[q4 * 4 + 1] = (v.y - myMu) * myRs * s_gamma[e + 1] + s_beta[e + 1];
            hv[q4 * 4 + 2] = (v.z - myMu) * myRs * s_gamma[e + 2] + s_beta[e + 2];
            hv[q4 * 4 + 3] = (v.w - myMu) * myRs * s_gamma[e + 3] + s_beta[e + 3];
        }
        FragU hw;
        hw.bf = __builtin_convertvector(hv, v16bf);   // packed f32->bf16
        *(u32x4*)&s_h[wave][myTok][dh + 0] = hw.q[0];
        *(u32x4*)&s_h[wave][myTok][dh + 8] = hw.q[1];
        // (wave-private tile + in-order DS pipe: no barrier required)

        // A fragment for this K-chunk
        FragU a;
        a.q[0] = *(const u32x4*)&s_h[wave][arow][aoff];
        a.q[1] = *(const u32x4*)&s_h[wave][arow][aoff + 8];

        #pragma unroll
        for (int nT = 0; nT < NTILE_DOWN; ++nT) {
            FragU b;
            const u32x4* bp =
                (const u32x4*)(Bd + (((size_t)(kc * NTILE_DOWN + nT)) * 32 + lane) * 16);
            b.q[0] = bp[0];
            b.q[1] = bp[1];
            accD[nT] = __builtin_amdgcn_wmma_f32_16x16x32_bf16(
                false, a.bf, false, b.bf, (short)0, accD[nT], false, false);
        }
    }

    // bias + ReLU, write z tile (bf16) to LDS
    const int cLane = lane & 15;
    const int tb    = (lane < 16) ? 0 : 8;   // C-layout row base
    #pragma unroll
    for (int nT = 0; nT < NTILE_DOWN; ++nT) {
        const int c = nT * 16 + cLane;
        const float bd = b_down[c];
        v8f zf;
        #pragma unroll
        for (int i = 0; i < 8; ++i)
            zf[i] = fmaxf(accD[nT][i] + bd, 0.f);
        Bf8U zc;
        zc.bf = __builtin_convertvector(zf, v8bf);    // packed f32->bf16
        #pragma unroll
        for (int i = 0; i < 8; ++i)
            s_z[wave][tb + i][c] = zc.s[i];
    }

    // ---------------- Phase 3: up-proj WMMA + epilogue ----------------
    FragU zA[KFRAG_UP];
    #pragma unroll
    for (int kT = 0; kT < KFRAG_UP; ++kT) {
        zA[kT].q[0] = *(const u32x4*)&s_z[wave][arow][kT * 32 + aoff];
        zA[kT].q[1] = *(const u32x4*)&s_z[wave][arow][kT * 32 + aoff + 8];
    }

    float* ow = out + (size_t)tokBase * DIMD;
    for (int nT = 0; nT < NTILE_UP; ++nT) {
        v8f acc = {};
        #pragma unroll
        for (int kT = 0; kT < KFRAG_UP; ++kT) {
            FragU b;
            const u32x4* bp =
                (const u32x4*)(Bu + (((size_t)(kT * NTILE_UP + nT)) * 32 + lane) * 16);
            b.q[0] = bp[0];
            b.q[1] = bp[1];
            acc = __builtin_amdgcn_wmma_f32_16x16x32_bf16(
                false, zA[kT].bf, false, b.bf, (short)0, acc, false, false);
        }
        const int c = nT * 16 + cLane;
        const float bu = b_up[c];
        #pragma unroll
        for (int i = 0; i < 8; ++i)
            ow[(size_t)(tb + i) * DIMD + c] = (acc[i] + bu) * SCALE;
    }
}

// ---------------------------------------------------------------------------
extern "C" void kernel_launch(void* const* d_in, const int* in_sizes, int n_in,
                              void* d_out, int out_size, void* d_ws, size_t ws_size,
                              hipStream_t stream) {
    (void)n_in; (void)out_size; (void)ws_size;
    const float* x     = (const float*)d_in[0];
    const float* gamma = (const float*)d_in[1];
    const float* beta  = (const float*)d_in[2];
    const float* Wd    = (const float*)d_in[3];
    const float* bd    = (const float*)d_in[4];
    const float* Wu    = (const float*)d_in[5];
    const float* bu    = (const float*)d_in[6];
    float* out = (float*)d_out;

    const int ntok = in_sizes[0] / DIMD;

    unsigned short* Bd = (unsigned short*)d_ws;                    // 96 KiB
    unsigned short* Bu = Bd + (size_t)KFRAG_DOWN * NTILE_DOWN * 32 * 16;  // +96 KiB

    pack_weights_kernel<<<KFRAG_DOWN * NTILE_DOWN + KFRAG_UP * NTILE_UP, 32, 0, stream>>>(
        Wd, Wu, Bd, Bu);

    const int blocks = (ntok + TOK_PER_BLOCK - 1) / TOK_PER_BLOCK;
    adapter_kernel<<<blocks, 256, 0, stream>>>(x, gamma, beta, bd, bu, Bd, Bu, out, ntok);
}
